// MultiRangeAttention_53764400611372
// MI455X (gfx1250) — matrix operand order repaired and checked by reading, not compiled, not run on hardware
//
#include <hip/hip_runtime.h>
#include <hip/hip_bf16.h>

typedef _Float16 half_t;
typedef __attribute__((ext_vector_type(16))) _Float16 v16h;
typedef __attribute__((ext_vector_type(8)))  _Float16 v8h;
typedef __attribute__((ext_vector_type(8)))  float    v8f;

union V16 { v16h v; v8h h[2]; };

__device__ __forceinline__ v8f wmma_f32_f16(v16h a, v16h b, v8f c) {
    // D = A(16x32 f16) * B(32x16 f16) + C(16x16 f32)
    return __builtin_amdgcn_wmma_f32_16x16x32_f16(
        /*neg_a=*/false, a, /*neg_b=*/false, b,
        /*c_mod=*/(short)0, c, /*reuse_a=*/false, /*reuse_b=*/false);
}

// ---------------------------------------------------------------------------
// Kernel 0: one-shot f32 -> f16 weight conversion (row-major preserved).
// Weights are L2-resident; converting once removes 16 loads + 16 cvts per
// K-step from every GEMM wave (they become two global_load_b128).
// ---------------------------------------------------------------------------
__global__ __launch_bounds__(256) void cvt_f16_kernel(
    const float* __restrict__ src, half_t* __restrict__ dst, int n)
{
    const int i = (blockIdx.x * 256 + threadIdx.x) * 4;
    if (i + 3 < n) {
#pragma unroll
        for (int j = 0; j < 4; ++j) dst[i + j] = (half_t)src[i + j];
    }
}

// ---------------------------------------------------------------------------
// Kernel 1: qkv = W_qkv (768x256) @ x[b] (256x1024) + bias
//   writes q,k as f32 (b,c,n) for depthwise conv; v directly as f16 (b,c,n).
// Block: 256 thr (8 waves), one 16-wide n-tile per block; each wave does
// 6 oc-tiles x 8 K-steps = 48 WMMAs. x tile staged transposed in LDS.
// ---------------------------------------------------------------------------
__global__ __launch_bounds__(256) void qkv_gemm_kernel(
    const float* __restrict__ x, const half_t* __restrict__ w16,
    const float* __restrict__ bias, float* __restrict__ qf,
    float* __restrict__ kf, half_t* __restrict__ vh)
{
    __shared__ half_t xs[16][264];          // [n_local][c], padded rows (16B aligned)
    const int tid = threadIdx.x;
    const int n0  = blockIdx.x * 16;
    const int b   = blockIdx.y;

    {   // stage x^T tile: thread t owns channel c=t, scatters 16 n values
        const int c = tid;
        const float* xp = x + ((size_t)b * 256 + c) * 1024 + n0;
#pragma unroll
        for (int j = 0; j < 16; ++j) xs[j][c] = (half_t)xp[j];
    }
    __syncthreads();

    const int wave = tid >> 5, lane = tid & 31;
    const bool hi  = lane >= 16;
    const int l15  = lane & 15;
    const int kb   = hi ? 8 : 0;    // A halves K base
    const int cbB  = hi ? 16 : 0;   // unused; B uses kb as A does

    for (int t = 0; t < 6; ++t) {
        const int oc0 = (t * 8 + wave) * 16;
        const half_t* wrow = w16 + (size_t)(oc0 + l15) * 256;
        v8f acc = {};
        for (int c0 = 0; c0 < 256; c0 += 32) {
            V16 A;
            A.h[0] = *(const v8h*)(wrow + c0 + kb);
            A.h[1] = *(const v8h*)(wrow + c0 + 16 + kb);
            V16 Bm;
            Bm.h[0] = *(const v8h*)&xs[l15][c0 + kb];
            Bm.h[1] = *(const v8h*)&xs[l15][c0 + 16 + kb];
            acc = wmma_f32_f16(A.v, Bm.v, acc);
        }
#pragma unroll
        for (int r = 0; r < 8; ++r) {
            const int row = oc0 + r + (hi ? 8 : 0);   // output channel 0..767
            const float val = acc[r] + bias[row];
            const int n = n0 + l15;
            if (row < 256)       qf[((size_t)b * 256 + row)        * 1024 + n] = val;
            else if (row < 512)  kf[((size_t)b * 256 + (row - 256)) * 1024 + n] = val;
            else                 vh[((size_t)b * 256 + (row - 512)) * 1024 + n] = (half_t)val;
        }
    }
}

// ---------------------------------------------------------------------------
// Kernel 2: multi-scale depthwise conv + residual (+ q pre-scale).
// Channel groups of 64 with k = 1,3,5,7 (= 2g+1). Output re-laid-out to
// f16 (bh, n, d) [d contiguous] — the WMMA A/B-friendly layout for attention.
// q is pre-scaled by HEAD_DIM^-0.5 * log2(e) so attention can use exp2.
// ---------------------------------------------------------------------------
__global__ __launch_bounds__(256) void msdw_kernel(
    const float* __restrict__ src,
    const float* __restrict__ w0, const float* __restrict__ w1,
    const float* __restrict__ w2, const float* __restrict__ w3,
    const float* __restrict__ b0, const float* __restrict__ b1,
    const float* __restrict__ b2, const float* __restrict__ b3,
    float scale, half_t* __restrict__ dst)
{
    const int p = blockIdx.x * 256 + threadIdx.x;   // pixel 0..1023
    const int c = blockIdx.y;                       // channel 0..255
    const int b = blockIdx.z;                       // batch
    const int g = c >> 6;                           // group -> k = 2g+1
    const float* wg; const float* bg;
    if (g == 0)      { wg = w0; bg = b0; }
    else if (g == 1) { wg = w1; bg = b1; }
    else if (g == 2) { wg = w2; bg = b2; }
    else             { wg = w3; bg = b3; }
    const int ks = 2 * g + 1;
    const int lc = c & 63;
    const float* wp = wg + (size_t)lc * ks * ks;
    const int px = p & 31, py = p >> 5;
    const float* sp = src + ((size_t)b * 256 + c) * 1024;
    float acc = bg[lc];
    for (int dy = -g; dy <= g; ++dy) {
        const int yy = py + dy;
        if (yy < 0 || yy >= 32) continue;
        for (int dx = -g; dx <= g; ++dx) {
            const int xx = px + dx;
            if (xx < 0 || xx >= 32) continue;
            acc += sp[yy * 32 + xx] * wp[(dy + g) * ks + (dx + g)];
        }
    }
    const float val = (sp[p] + acc) * scale;        // residual + msdw
    // layout: (b, h, n, d) with d contiguous; c = h*32 + d
    dst[(((size_t)b * 8 + (c >> 5)) * 1024 + p) * 32 + (c & 31)] = (half_t)val;
}

// ---------------------------------------------------------------------------
// Kernel 3: flash attention. One wave = 16 query rows; grid (8 q-blocks, 64 bh).
// HEAD_DIM = 32 = WMMA K-dim: S tile = 1 WMMA; PV = 2 WMMAs per 32-key block.
// Online softmax in base-2 domain (scale*log2e folded into q) via exp2f,
// row reductions with __shfl_xor inside 16-lane halves (matches C/D layout).
// P re-layout D->A goes through a per-wave LDS staging tile (in-order DS).
// ---------------------------------------------------------------------------
__global__ __launch_bounds__(256) void attn_kernel(
    const half_t* __restrict__ q16, const half_t* __restrict__ k16,
    const half_t* __restrict__ v16, half_t* __restrict__ ao)
{
    __shared__ half_t pst[8][16 * 32];              // per-wave P staging (1 KiB each)
    const int tid  = threadIdx.x;
    const int wave = tid >> 5, lane = tid & 31;
    const bool hi  = lane >= 16;
    const int l15  = lane & 15;
    const int bh = blockIdx.y, b = bh >> 3, h = bh & 7;
    const int q0 = blockIdx.x * 128 + wave * 16;
    const int kb      = hi ? 8 : 0;                 // A half-K base
    const int halfsel = hi ? 16 : 0;                // unused; B uses kb as A does
    const int rowoff  = hi ? 8 : 0;                 // C/D row offset

    V16 A;                                          // Q tile, loaded once
    {   const half_t* qr = q16 + ((size_t)bh * 1024 + q0 + l15) * 32;
        A.h[0] = *(const v8h*)(qr + kb);
        A.h[1] = *(const v8h*)(qr + 16 + kb);
    }

    v8f O0 = {}, O1 = {};                           // output acc, d=0..15 / 16..31
    float mrun[8], lrun[8];
#pragma unroll
    for (int r = 0; r < 8; ++r) { mrun[r] = -3.0e38f; lrun[r] = 0.f; }

    const half_t* kbp = k16 + (size_t)bh * 1024 * 32;            // (n, d)
    const half_t* vbp = v16 + ((size_t)b * 256 + h * 32) * 1024; // (d, n)
    half_t* ps = &pst[wave][0];

    for (int n0 = 0; n0 < 1024; n0 += 32) {
        // --- S = Q K^T for 32 keys: two 16-key B tiles ---
        V16 B0, B1;
        const half_t* kr0 = kbp + (size_t)(n0 + l15) * 32;
        const half_t* kr1 = kr0 + 16 * 32;
        B0.h[0] = *(const v8h*)(kr0 + kb); B0.h[1] = *(const v8h*)(kr0 + 16 + kb);
        B1.h[0] = *(const v8h*)(kr1 + kb); B1.h[1] = *(const v8h*)(kr1 + 16 + kb);
        v8f S0 = {}, S1 = {};
        S0 = wmma_f32_f16(A.v, B0.v, S0);
        S1 = wmma_f32_f16(A.v, B1.v, S1);

        if (n0 + 32 < 1024) __builtin_prefetch(kr0 + 32 * 32, 0, 1);

        // --- online softmax update (base-2 domain; row in one 16-lane half) ---
        v8f P0, P1;
#pragma unroll
        for (int r = 0; r < 8; ++r) {
            float mx = fmaxf(S0[r], S1[r]);
            mx = fmaxf(mx, __shfl_xor(mx, 1));
            mx = fmaxf(mx, __shfl_xor(mx, 2));
            mx = fmaxf(mx, __shfl_xor(mx, 4));
            mx = fmaxf(mx, __shfl_xor(mx, 8));
            const float mn   = fmaxf(mrun[r], mx);
            const float corr = exp2f(mrun[r] - mn);
            mrun[r] = mn;
            const float p0 = exp2f(S0[r] - mn);
            const float p1 = exp2f(S1[r] - mn);
            P0[r] = p0; P1[r] = p1;
            float s = p0 + p1;
            s += __shfl_xor(s, 1);
            s += __shfl_xor(s, 2);
            s += __shfl_xor(s, 4);
            s += __shfl_xor(s, 8);
            lrun[r] = lrun[r] * corr + s;
            O0[r] *= corr; O1[r] *= corr;
        }

        // --- D-layout f32 -> A-layout f16 via wave-private LDS tile ---
#pragma unroll
        for (int r = 0; r < 8; ++r) {
            ps[(r + rowoff) * 32 + l15]      = (half_t)P0[r];
            ps[(r + rowoff) * 32 + 16 + l15] = (half_t)P1[r];
        }
        V16 PA;
        {   const half_t* pr = ps + l15 * 32;
            PA.h[0] = *(const v8h*)(pr + kb);
            PA.h[1] = *(const v8h*)(pr + 16 + kb);
        }

        // --- O += P @ V^T: B[K=key][N=d] = v(d,n) contiguous along n ---
        V16 BV0, BV1;
        const half_t* vr0 = vbp + (size_t)l15 * 1024 + n0;
        const half_t* vr1 = vr0 + 16 * 1024;
        BV0.h[0] = *(const v8h*)(vr0 + kb); BV0.h[1] = *(const v8h*)(vr0 + 16 + kb);
        BV1.h[0] = *(const v8h*)(vr1 + kb); BV1.h[1] = *(const v8h*)(vr1 + 16 + kb);
        O0 = wmma_f32_f16(PA.v, BV0.v, O0);
        O1 = wmma_f32_f16(PA.v, BV1.v, O1);
    }

    // --- normalize and store as (b, n, c) f16, c contiguous (proj A layout) ---
#pragma unroll
    for (int r = 0; r < 8; ++r) {
        const float inv = 1.f / lrun[r];
        const int n = q0 + r + rowoff;
        half_t* op = ao + ((size_t)b * 1024 + n) * 256 + h * 32;
        op[l15]      = (half_t)(O0[r] * inv);
        op[16 + l15] = (half_t)(O1[r] * inv);
    }
}

// ---------------------------------------------------------------------------
// Kernel 4: out[b,o,n] = sum_c ao[b,n,c] * W[o,c] + bias[o].
// A = attn-out (n x c) f16, B = pre-converted proj_w f16 rows (contiguous).
// ---------------------------------------------------------------------------
__global__ __launch_bounds__(256) void proj_kernel(
    const half_t* __restrict__ ao, const half_t* __restrict__ w16,
    const float* __restrict__ bias, float* __restrict__ out)
{
    const int tid  = threadIdx.x;
    const int wave = tid >> 5, lane = tid & 31;
    const bool hi  = lane >= 16;
    const int l15  = lane & 15;
    const int n0 = blockIdx.x * 16;
    const int o0 = (blockIdx.y * 8 + wave) * 16;
    const int b  = blockIdx.z;
    const int kb  = hi ? 8 : 0;
    const int cbB = hi ? 16 : 0;

    const half_t* arow = ao + ((size_t)b * 1024 + n0 + l15) * 256;
    const half_t* wrow = w16 + (size_t)(o0 + l15) * 256;
    v8f acc = {};
    for (int c0 = 0; c0 < 256; c0 += 32) {
        V16 A;
        A.h[0] = *(const v8h*)(arow + c0 + kb);
        A.h[1] = *(const v8h*)(arow + c0 + 16 + kb);
        V16 Bm;
        Bm.h[0] = *(const v8h*)(wrow + c0 + kb);
        Bm.h[1] = *(const v8h*)(wrow + c0 + 16 + kb);
        acc = wmma_f32_f16(A.v, Bm.v, acc);
    }
    const float bb = bias[o0 + l15];
#pragma unroll
    for (int r = 0; r < 8; ++r) {
        const int n = n0 + r + (hi ? 8 : 0);
        out[((size_t)b * 256 + (o0 + l15)) * 1024 + n] = acc[r] + bb;
    }
}

// ---------------------------------------------------------------------------
// Host launcher. Input order (tuples flattened in dict order):
//  0:x 1:qkv_w 2:qkv_b 3-6:qdw_w 7-10:qdw_b 11-14:kdw_w 15-18:kdw_b
//  19:proj_w 20:proj_b
// Workspace map (~32.5 MiB, fully rewritten each call):
//  [0,8M) q f32 | [8M,16M) k f32 | [16M,20M) v f16 | [20M,24M) q16 |
//  [24M,28M) k16 | [28M,32M) attn-out f16 | [32M,32M+384K) qkv_w f16 |
//  [32M+512K, +128K) proj_w f16
// ---------------------------------------------------------------------------
extern "C" void kernel_launch(void* const* d_in, const int* in_sizes, int n_in,
                              void* d_out, int out_size, void* d_ws, size_t ws_size,
                              hipStream_t stream) {
    const float* x      = (const float*)d_in[0];
    const float* qkv_w  = (const float*)d_in[1];
    const float* qkv_b  = (const float*)d_in[2];
    const float* qw0 = (const float*)d_in[3],  *qw1 = (const float*)d_in[4];
    const float* qw2 = (const float*)d_in[5],  *qw3 = (const float*)d_in[6];
    const float* qb0 = (const float*)d_in[7],  *qb1 = (const float*)d_in[8];
    const float* qb2 = (const float*)d_in[9],  *qb3 = (const float*)d_in[10];
    const float* kw0 = (const float*)d_in[11], *kw1 = (const float*)d_in[12];
    const float* kw2 = (const float*)d_in[13], *kw3 = (const float*)d_in[14];
    const float* kb0 = (const float*)d_in[15], *kb1 = (const float*)d_in[16];
    const float* kb2 = (const float*)d_in[17], *kb3 = (const float*)d_in[18];
    const float* proj_w = (const float*)d_in[19];
    const float* proj_b = (const float*)d_in[20];

    char* ws = (char*)d_ws;
    float*  qf   = (float*)(ws);
    float*  kf   = (float*)(ws + ((size_t)8  << 20));
    half_t* vh   = (half_t*)(ws + ((size_t)16 << 20));
    half_t* q16  = (half_t*)(ws + ((size_t)20 << 20));
    half_t* k16  = (half_t*)(ws + ((size_t)24 << 20));
    half_t* ao   = (half_t*)(ws + ((size_t)28 << 20));
    half_t* wq16 = (half_t*)(ws + ((size_t)32 << 20));
    half_t* wp16 = (half_t*)(ws + ((size_t)32 << 20) + (512u << 10));

    // one-shot weight conversion (tiny; L2-resident afterwards)
    cvt_f16_kernel<<<dim3(192), 256, 0, stream>>>(qkv_w, wq16, 768 * 256);
    cvt_f16_kernel<<<dim3(64),  256, 0, stream>>>(proj_w, wp16, 256 * 256);

    qkv_gemm_kernel<<<dim3(64, 8), 256, 0, stream>>>(x, wq16, qkv_b, qf, kf, vh);

    // q pre-scale: HEAD_DIM^-0.5 * log2(e) -> softmax runs in base-2 domain
    const float qscale = 0.17677669529663687f * 1.44269504088896340f;
    msdw_kernel<<<dim3(4, 256, 8), 256, 0, stream>>>(
        qf, qw0, qw1, qw2, qw3, qb0, qb1, qb2, qb3, qscale, q16);
    msdw_kernel<<<dim3(4, 256, 8), 256, 0, stream>>>(
        kf, kw0, kw1, kw2, kw3, kb0, kb1, kb2, kb3, 1.0f, k16);

    attn_kernel<<<dim3(8, 64), 256, 0, stream>>>(q16, k16, vh, ao);

    proj_kernel<<<dim3(64, 2, 8), 256, 0, stream>>>(ao, wp16, proj_b, (float*)d_out);
}
